// GCNWithEmbeddings_91044716740867
// MI455X (gfx1250) — compile-verified, matrix operand order checked by reading, not run on hardware
//
#include <hip/hip_runtime.h>
#include <hip/hip_bf16.h>

// ---------------- types ----------------
typedef __attribute__((ext_vector_type(16))) __bf16 v16bf;
typedef __attribute__((ext_vector_type(8)))  float  v8f;

// Exact RNE (used only in one-time weight swizzle).
__device__ __forceinline__ unsigned short f2bf_rne(float f) {
    unsigned u = __float_as_uint(f);
    u += 0x7fffu + ((u >> 16) & 1u);
    return (unsigned short)(u >> 16);
}
// Fast round-half-up bf16 (activation path): 2 VALU.
__device__ __forceinline__ unsigned short f2bf(float f) {
    return (unsigned short)((__float_as_uint(f) + 0x8000u) >> 16);
}
// Pack two floats to packed bf16x2 with one v_perm_b32: 3 VALU per pair.
__device__ __forceinline__ unsigned pack2_bf16(float a, float b) {
    unsigned ua = __float_as_uint(a) + 0x8000u;
    unsigned ub = __float_as_uint(b) + 0x8000u;
    return __builtin_amdgcn_perm(ub, ua, 0x07060302u);  // {ub[31:16], ua[31:16]}
}

__device__ __forceinline__ v8f wmma_bf16(v16bf a, v16bf b, v8f c) {
    return __builtin_amdgcn_wmma_f32_16x16x32_bf16(
        false, a, false, b, (short)0, c, false, false);
}

// A fragment from row-major LDS [M][Kpad] bf16.
__device__ __forceinline__ v16bf load_frag_lds(const unsigned short* base, int m,
                                               int kpad, int koff) {
    const unsigned short* p = base + m * kpad + koff;
    union { uint4 q[2]; v16bf v; } u;
    u.q[0] = *(const uint4*)(p);
    u.q[1] = *(const uint4*)(p + 16);
    return u.v;
}
// B fragment from LDS fragment buffer (16 contiguous bf16 per lane per tile).
__device__ __forceinline__ v16bf load_fragB_lds(const unsigned short* base, int tile,
                                                int lane) {
    const uint4* p = (const uint4*)(base + ((size_t)tile * 512 + lane * 16));
    union { uint4 q[2]; v16bf v; } u;
    u.q[0] = p[0];
    u.q[1] = p[1];
    return u.v;
}
// B fragment from global pre-swizzled weights.
__device__ __forceinline__ v16bf load_frag_global(const unsigned short* swz,
                                                  int tile, int lane) {
    const uint4* p = (const uint4*)(swz + ((size_t)tile * 32 + lane) * 16);
    union { uint4 q[2]; v16bf v; } u;
    u.q[0] = p[0];
    u.q[1] = p[1];
    return u.v;
}

// ---------------- weight pre-swizzle ----------------
// W row-major [Nout, K] fp32 -> bf16 B-fragments for B = W^T. tile = kt*ntiles + nt.
__global__ void swizzleB(const float* __restrict__ W, unsigned short* __restrict__ out,
                         int K, int ntiles, int ktiles) {
    int tid = blockIdx.x * blockDim.x + threadIdx.x;
    int total = ktiles * ntiles * 32;
    if (tid >= total) return;
    int lane = tid & 31;
    int tile = tid >> 5;
    int kt = tile / ntiles;
    int nt = tile - kt * ntiles;
    int n = nt * 16 + (lane & 15);
    int koff0 = (lane >> 4) * 8;

    union { unsigned short s[16]; uint4 q[2]; } o;
#pragma unroll
    for (int r = 0; r < 8; ++r) {
#pragma unroll
        for (int p = 0; p < 2; ++p) {
            int kl = (r < 4) ? (koff0 + 2 * r + p) : (16 + koff0 + 2 * (r - 4) + p);
            o.s[2 * r + p] = f2bf_rne(W[(size_t)n * K + kt * 32 + kl]);
        }
    }
    uint4* dst = (uint4*)(out + (size_t)tile * 512 + lane * 16);
    dst[0] = o.q[0];
    dst[1] = o.q[1];
}

// ---------------- fused per-graph kernel ----------------
// B=512, S=96, N=64, H=512, E=512 (+64 self-loops).
#define KP1 104   // x^T    [64 x 96 ] bf16
#define KP2 520   // A2     [64 x 512] bf16
#define KP3 200   // im2col [96 x 192] bf16
#define KPA 72    // Ahat   [64 x 64 ] bf16
#define GP  104   // g      [64 x 96 ] fp32

#define R1_OFF 65536            // A region (66560 B)
#define R2_OFF 132096           // Ahat bf16 (9216 B)
#define R3_OFF 141312           // meta (512 B)
#define SMEM_BYTES 141824

__global__ __launch_bounds__(256)
void gcn_fused(const float* __restrict__ x, const int* __restrict__ ei,
               const float* __restrict__ b1, const float* __restrict__ b2,
               const unsigned short* __restrict__ w1s,
               const unsigned short* __restrict__ w2s,
               const unsigned short* __restrict__ wcs,
               float* __restrict__ out) {
    __shared__ __align__(16) unsigned char smem[SMEM_BYTES];
    unsigned short* fragB = (unsigned short*)smem;            // h0B (64KB) / h1B (12KB)
    float*          AhatF = (float*)smem;                     // stage-0 alias (16KB)
    float*          g     = (float*)(smem + 16384);           // fp32 [64][104]
    unsigned short* areg  = (unsigned short*)(smem + R1_OFF); // A1 / A2 / A3
    unsigned short* ahat  = (unsigned short*)(smem + R2_OFF); // bf16 [64][72]
    int*   deg  = (int*)(smem + R3_OFF);
    float* dinv = (float*)(smem + R3_OFF + 256);

    const int tid  = threadIdx.x;
    const int lane = tid & 31;
    const int wave = tid >> 5;
    const int b    = blockIdx.x;

    const int* srcp = ei + (size_t)b * 1024;
    const int* dstp = srcp + 512;

    // ---- stage 0: x^T -> LDS ; dense normalized adjacency Ahat ----
    if (tid < 64) deg[tid] = 0;
    for (int t = tid; t < 4096; t += 256) AhatF[t] = 0.0f;
    __syncthreads();

    for (int t = tid; t < 64 * 96; t += 256) {
        int n = t & 63, s = t >> 6;                          // coalesced over n
        areg[n * KP1 + s] = f2bf(x[((size_t)b * 96 + s) * 64 + n]);
    }
    for (int e = tid; e < 576; e += 256) {
        int d = (e < 512) ? dstp[e] : (e - 512);
        atomicAdd(&deg[d], 1);
    }
    __syncthreads();
    if (tid < 64) dinv[tid] = rsqrtf((float)deg[tid]);       // deg>=1 (self-loop)
    __syncthreads();
    for (int e = tid; e < 576; e += 256) {
        int s, d;
        if (e < 512) { s = srcp[e]; d = dstp[e]; }
        else         { s = d = e - 512; }
        atomicAdd(&AhatF[d * 64 + s], dinv[s] * dinv[d]);    // handles multi-edges
    }
    __syncthreads();
    for (int t = tid; t < 4096; t += 256)
        ahat[(t >> 6) * KPA + (t & 63)] = f2bf(AhatF[t]);
    __syncthreads();

    // ---- GEMM1: h0 = A1[64,96] x W1^T -> stored directly in B-fragment layout ----
    {
        const int mtile = wave >> 1;
        const int ntb   = (wave & 1) * 16;
        const int m     = mtile * 16 + (lane & 15);
        const int kg8   = (lane >> 4) * 8;
        const int kt_o  = mtile >> 1;      // dest B-fragment k-tile
        const int half  = mtile & 1;       // dest K-half within fragment
        v16bf a0 = load_frag_lds(areg, m, KP1, 0 * 32 + kg8);
        v16bf a1 = load_frag_lds(areg, m, KP1, 1 * 32 + kg8);
        v16bf a2 = load_frag_lds(areg, m, KP1, 2 * 32 + kg8);
        for (int nt = 0; nt < 16; ++nt) {
            int ntile = ntb + nt;
            v8f c = {};
            c = wmma_bf16(a0, load_frag_global(w1s, 0 * 32 + ntile, lane), c);
            c = wmma_bf16(a1, load_frag_global(w1s, 1 * 32 + ntile, lane), c);
            c = wmma_bf16(a2, load_frag_global(w1s, 2 * 32 + ntile, lane), c);
            uint4 pk;
            pk.x = pack2_bf16(c[0], c[1]);
            pk.y = pack2_bf16(c[2], c[3]);
            pk.z = pack2_bf16(c[4], c[5]);
            pk.w = pack2_bf16(c[6], c[7]);
            *(uint4*)(fragB + ((kt_o * 32 + ntile) * 512 + lane * 16 + half * 8)) = pk;
        }
    }
    __syncthreads();

    // ---- Aggregation 1 as GEMM: A2 = ReLU(Ahat[64,64] x h0 + b1) -> row-major bf16 ----
    {
        const int mtile = wave >> 1;
        const int ntb   = (wave & 1) * 16;
        const int m     = mtile * 16 + (lane & 15);
        const int kg8   = (lane >> 4) * 8;
        v16bf a0 = load_frag_lds(ahat, m, KPA, 0 * 32 + kg8);
        v16bf a1 = load_frag_lds(ahat, m, KPA, 1 * 32 + kg8);
        for (int nt = 0; nt < 16; ++nt) {
            int ntile = ntb + nt;
            v8f c = {};
            c = wmma_bf16(a0, load_fragB_lds(fragB, 0 * 32 + ntile, lane), c);
            c = wmma_bf16(a1, load_fragB_lds(fragB, 1 * 32 + ntile, lane), c);
            int row0 = mtile * 16 + 8 * (lane >> 4);
            int col  = ntile * 16 + (lane & 15);
            float bb = b1[col];
#pragma unroll
            for (int r = 0; r < 8; ++r)
                areg[(row0 + r) * KP2 + col] = f2bf(fmaxf(c[r] + bb, 0.0f));
        }
    }
    __syncthreads();

    // ---- GEMM2: h1 = A2[64,512] x W2^T -> B-fragment layout.
    //      One shared A fragment per kt feeds 3 independent accumulators. ----
    {
        const int mtile = wave >> 1;             // 4 mtiles over 8 waves
        const int ntb   = (wave & 1) * 3;        // 6 ntiles
        const int mm    = mtile * 16 + (lane & 15);
        const int kg8   = (lane >> 4) * 8;
        v8f c0 = {}, c1 = {}, c2 = {};
        for (int kt = 0; kt < 16; ++kt) {
            v16bf a = load_frag_lds(areg, mm, KP2, kt * 32 + kg8);
            c0 = wmma_bf16(a, load_frag_global(w2s, kt * 6 + ntb + 0, lane), c0);
            c1 = wmma_bf16(a, load_frag_global(w2s, kt * 6 + ntb + 1, lane), c1);
            c2 = wmma_bf16(a, load_frag_global(w2s, kt * 6 + ntb + 2, lane), c2);
        }
        const int kt_o = mtile >> 1, half = mtile & 1;
        v8f cc[3] = {c0, c1, c2};
#pragma unroll
        for (int i = 0; i < 3; ++i) {
            uint4 pk;
            pk.x = pack2_bf16(cc[i][0], cc[i][1]);
            pk.y = pack2_bf16(cc[i][2], cc[i][3]);
            pk.z = pack2_bf16(cc[i][4], cc[i][5]);
            pk.w = pack2_bf16(cc[i][6], cc[i][7]);
            *(uint4*)(fragB + ((kt_o * 6 + ntb + i) * 512 + lane * 16 + half * 8)) = pk;
        }
    }
    __syncthreads();

    // ---- Aggregation 2 as GEMM: g = Ahat x h1 + b2  (fp32 row-major) ----
    {
        const int mtile = wave >> 1;
        const int ntb   = (wave & 1) * 3;
        const int m     = mtile * 16 + (lane & 15);
        const int kg8   = (lane >> 4) * 8;
        v16bf a0 = load_frag_lds(ahat, m, KPA, 0 * 32 + kg8);
        v16bf a1 = load_frag_lds(ahat, m, KPA, 1 * 32 + kg8);
        for (int i = 0; i < 3; ++i) {
            int ntile = ntb + i;
            v8f c = {};
            c = wmma_bf16(a0, load_fragB_lds(fragB, 0 * 6 + ntile, lane), c);
            c = wmma_bf16(a1, load_fragB_lds(fragB, 1 * 6 + ntile, lane), c);
            int row0 = mtile * 16 + 8 * (lane >> 4);
            int col  = ntile * 16 + (lane & 15);
            float bb = b2[col];
#pragma unroll
            for (int r = 0; r < 8; ++r)
                g[(row0 + r) * GP + col] = c[r] + bb;
        }
    }
    __syncthreads();

    // ---- Circular im2col: A3[s][n*3+k] = g[n][(s+k-1) mod 96] (bf16) ----
    for (int t = tid; t < 64 * 96; t += 256) {
        int n = t / 96, s = t - n * 96;
        const float* gr = g + n * GP;
        int sm = (s == 0) ? 95 : s - 1;
        int sp = (s == 95) ? 0 : s + 1;
        unsigned short* dst3 = areg + s * KP3 + n * 3;
        dst3[0] = f2bf(gr[sm]);
        dst3[1] = f2bf(gr[s]);
        dst3[2] = f2bf(gr[sp]);
    }
    __syncthreads();

    // ---- GEMM3: out[b][96,512] = A3[96,192] x Wc^T.
    //      Each wave: 4 ntiles x all 6 mtiles; shared A per kt, 4 accumulators. ----
    {
        const int ntb = wave * 4;
        const int kg8 = (lane >> 4) * 8;
        for (int mtile = 0; mtile < 6; ++mtile) {
            const int mm = mtile * 16 + (lane & 15);
            v8f c0 = {}, c1 = {}, c2 = {}, c3 = {};
            for (int kt = 0; kt < 6; ++kt) {
                v16bf a = load_frag_lds(areg, mm, KP3, kt * 32 + kg8);
                c0 = wmma_bf16(a, load_frag_global(wcs, kt * 32 + ntb + 0, lane), c0);
                c1 = wmma_bf16(a, load_frag_global(wcs, kt * 32 + ntb + 1, lane), c1);
                c2 = wmma_bf16(a, load_frag_global(wcs, kt * 32 + ntb + 2, lane), c2);
                c3 = wmma_bf16(a, load_frag_global(wcs, kt * 32 + ntb + 3, lane), c3);
            }
            int row0 = mtile * 16 + 8 * (lane >> 4);
            v8f cc[4] = {c0, c1, c2, c3};
#pragma unroll
            for (int i = 0; i < 4; ++i) {
                int col = (ntb + i) * 16 + (lane & 15);
                float* op = out + ((size_t)b * 96 + row0) * 512 + col;
#pragma unroll
                for (int r = 0; r < 8; ++r)
                    op[(size_t)r * 512] = cc[i][r];
            }
        }
    }
}

// ---------------- host launcher ----------------
extern "C" void kernel_launch(void* const* d_in, const int* in_sizes, int n_in,
                              void* d_out, int out_size, void* d_ws, size_t ws_size,
                              hipStream_t stream) {
    (void)in_sizes; (void)n_in; (void)out_size; (void)ws_size;
    const float* x     = (const float*)d_in[0];
    const int*   ei    = (const int*)d_in[1];
    const float* W1    = (const float*)d_in[2];   // [512, 96]
    const float* b1    = (const float*)d_in[3];   // [512]
    const float* W2    = (const float*)d_in[4];   // [96, 512]
    const float* b2    = (const float*)d_in[5];   // [96]
    const float* convw = (const float*)d_in[6];   // [512, 64, 3] == [512,192]
    float* out = (float*)d_out;

    unsigned short* w1s = (unsigned short*)d_ws;  // 3*32 tiles * 512 = 49152 elems
    unsigned short* w2s = w1s + 49152;            // 16*6 tiles * 512 = 49152 elems
    unsigned short* wcs = w2s + 49152;            // 6*32 tiles * 512 = 98304 elems

    {
        int t1 = 3 * 32 * 32;
        swizzleB<<<(t1 + 127) / 128, 128, 0, stream>>>(W1, w1s, 96, 32, 3);
        int t2 = 16 * 6 * 32;
        swizzleB<<<(t2 + 127) / 128, 128, 0, stream>>>(W2, w2s, 512, 6, 16);
        int t3 = 6 * 32 * 32;
        swizzleB<<<(t3 + 127) / 128, 128, 0, stream>>>(convw, wcs, 192, 32, 6);
    }

    gcn_fused<<<512, 256, 0, stream>>>(x, ei, b1, b2, w1s, w2s, wcs, out);
}
